// TestModel_45483703665345
// MI455X (gfx1250) — matrix operand order repaired and checked by reading, not compiled, not run on hardware
//
#include <hip/hip_runtime.h>
#include <hip/hip_bf16.h>

// ---------------- problem constants ----------------
#define NNODES 200000
#define EEDGES 400000
#define INF_   205
#define DDIM   200
#define DP     224          // padded feature dim: mult of 32 (K) and 16 (N)
#define NT_D   14           // DP/16 column tiles
#define KT     7            // DP/32 K tiles (also covers 205->224 for proj)
#define GATES_NT 42         // 3 gates * NT_D
#define GG     256

typedef __attribute__((ext_vector_type(16))) __bf16 v16bf;
typedef __attribute__((ext_vector_type(8)))  __bf16 v8bf;
typedef __attribute__((ext_vector_type(8)))  float  v8f;
typedef __attribute__((ext_vector_type(4)))  float  v4f;

// ---------------- WMMA fragment helpers ----------------
// A fragment (16x32 bf16, row-major source [rows, ld], ld mult of 16, fully padded):
// lane: m = lane&15, half = lane>>4 ; elements i<8 -> K = half*8+i ; i>=8 -> K = 16+half*8+(i-8)
__device__ __forceinline__ v16bf load_a_bf16(const __bf16* base, int ld,
                                             int rowBase, int kBase, int lane) {
    int m = lane & 15, half = lane >> 4;
    const __bf16* p = base + (size_t)(rowBase + m) * ld + kBase + half * 8;
    v8bf lo = *(const v8bf*)(p);
    v8bf hi = *(const v8bf*)(p + 16);
    v16bf a;
#pragma unroll
    for (int i = 0; i < 8; ++i) { a[i] = lo[i]; a[i + 8] = hi[i]; }
    return a;
}

// A fragment from fully padded f32 source (no bounds; 4x 16B vector loads + cvt)
__device__ __forceinline__ v16bf load_a_f32pad(const float* base, int ld,
                                               int rowBase, int kBase, int lane) {
    int m = lane & 15, half = lane >> 4;
    const float* p = base + (size_t)(rowBase + m) * ld + kBase + half * 8;
    v4f a0 = *(const v4f*)(p);
    v4f a1 = *(const v4f*)(p + 4);
    v4f b0 = *(const v4f*)(p + 16);
    v4f b1 = *(const v4f*)(p + 20);
    v16bf a;
#pragma unroll
    for (int i = 0; i < 4; ++i) {
        a[i]      = (__bf16)a0[i];
        a[i + 4]  = (__bf16)a1[i];
        a[i + 8]  = (__bf16)b0[i];
        a[i + 12] = (__bf16)b1[i];
    }
    return a;
}

// B fragment: pre-packed so each lane's 16 bf16 are contiguous (32B load)
__device__ __forceinline__ v16bf load_b(const __bf16* pack, int ktiles,
                                        int nt, int kt, int lane) {
    return *(const v16bf*)(pack + (((size_t)nt * ktiles + kt) * 32 + lane) * 16);
}

__device__ __forceinline__ v8f wmma_bf16(v16bf a, v16bf b, v8f c) {
    return __builtin_amdgcn_wmma_f32_16x16x32_bf16(false, a, false, b, (short)0, c, false, false);
}

// ---------------- weight packing ----------------
// dst[((nt*ktiles+kt)*32 + lane)*16 + i]  where k = kt*32 + (lane>>4)*8 + (i&7) + (i>>3)*16,
// n = nt*16 + (lane&15).
// gate_mode==0: src is [Ksrc, Nsrc] row-major (ld = Nsrc)
// gate_mode==1: src is [3*DDIM, DDIM] (torch weight), packed col n -> gate g=n/DP,
//               j=n%DP, element = src[(g*DDIM+j)*DDIM + k]
__global__ void pack_b_kernel(const float* __restrict__ src, __bf16* __restrict__ dst,
                              int ktiles, int ntiles, int Ksrc, int Nsrc, int gate_mode) {
    int id = blockIdx.x * blockDim.x + threadIdx.x;
    int total = ntiles * ktiles * 512;
    if (id >= total) return;
    int i    = id & 15;
    int lane = (id >> 4) & 31;
    int kt   = (id >> 9) % ktiles;
    int nt   = id / (512 * ktiles);
    int k = kt * 32 + ((lane >> 4) << 3) + (i & 7) + ((i >> 3) << 4);
    int n = nt * 16 + (lane & 15);
    float v = 0.0f;
    if (!gate_mode) {
        if (k < Ksrc && n < Nsrc) v = src[(size_t)k * Nsrc + n];
    } else {
        int g = n / DP, j = n % DP;
        if (k < DDIM && j < DDIM) v = src[((size_t)g * DDIM + j) * DDIM + k];
    }
    dst[id] = (__bf16)v;
}

// ---------------- fill ----------------
__global__ void fill_f32(float* __restrict__ p, float v, long n) {
    long i = (long)blockIdx.x * blockDim.x + threadIdx.x;
    if (i < n) p[i] = v;
}

// ---------------- x f32 [N,205] -> bf16 padded [N,224] ----------------
__global__ void convert_pad_x(const float* __restrict__ x, __bf16* __restrict__ xb) {
    int node = blockIdx.x * blockDim.y + threadIdx.y;
    if (node >= NNODES) return;
    const float* xp = x + (size_t)node * INF_;
    __bf16* op = xb + (size_t)node * DP;
#pragma unroll
    for (int j = 0; j < 7; ++j) {
        int f = threadIdx.x + j * 32;
        float v = (f < INF_) ? xp[f] : 0.0f;
        op[f] = (__bf16)v;
    }
}

// ---------------- proj GEMM: h = relu(xb @ proj_w + b), bf16 out [N, DP] ----------------
__global__ void proj_gemm(const __bf16* __restrict__ xb, const __bf16* __restrict__ packW,
                          const float* __restrict__ bias, __bf16* __restrict__ h) {
    int lane = threadIdx.x;
    int t = blockIdx.x * blockDim.y + threadIdx.y;
    int tiles = (NNODES / 16) * NT_D;
    if (t >= tiles) return;
    int mt = t / NT_D, nt = t % NT_D;
    v8f acc = {};
    if (nt < 13) {
#pragma unroll
        for (int kt = 0; kt < KT; ++kt) {
            v16bf a = load_a_bf16(xb, DP, mt * 16, kt * 32, lane);
            v16bf b = load_b(packW, KT, nt, kt, lane);
            acc = wmma_bf16(a, b, acc);
        }
    }
    int col = nt * 16 + (lane & 15);
    bool valid = (col < DDIM);
    int colc = valid ? col : (DDIM - 1);   // clamped (in-bounds) bias index
    float bcol = bias[colc];
#pragma unroll
    for (int r = 0; r < 8; ++r) {
        int row = mt * 16 + (lane >> 4) * 8 + r;
        float v = acc[r] + bcol;
        v = v > 0.0f ? v : 0.0f;
        v = valid ? v : 0.0f;              // branch-free select; pad cols -> 0
        h[(size_t)row * DP + col] = (__bf16)v;
    }
}

// ---------------- GGC GEMM: hw = h @ W, bf16 in/out ----------------
__global__ void ggc_gemm(const __bf16* __restrict__ h, const __bf16* __restrict__ packW,
                         __bf16* __restrict__ hw) {
    int lane = threadIdx.x;
    int t = blockIdx.x * blockDim.y + threadIdx.y;
    int tiles = (NNODES / 16) * NT_D;
    if (t >= tiles) return;
    int mt = t / NT_D, nt = t % NT_D;
    v8f acc = {};
    if (nt < 13) {
#pragma unroll
        for (int kt = 0; kt < KT; ++kt) {
            v16bf a = load_a_bf16(h, DP, mt * 16, kt * 32, lane);
            v16bf b = load_b(packW, KT, nt, kt, lane);
            acc = wmma_bf16(a, b, acc);
        }
    }
    int col = nt * 16 + (lane & 15);
    bool valid = (col < DDIM);
#pragma unroll
    for (int r = 0; r < 8; ++r) {
        int row = mt * 16 + (lane >> 4) * 8 + r;
        float v = valid ? acc[r] : 0.0f;
        hw[(size_t)row * DP + col] = (__bf16)v;
    }
}

// ---------------- edge scatter: m[dst] += hw[src] ----------------
__global__ void edge_scatter(const __bf16* __restrict__ hw, const int* __restrict__ ei,
                             float* __restrict__ m) {
    int e = blockIdx.x * blockDim.y + threadIdx.y;
    if (e >= EEDGES) return;
    int s = ei[e];
    int d = ei[EEDGES + e];
    const __bf16* hs = hw + (size_t)s * DP;
    float* md = m + (size_t)d * DP;
#pragma unroll
    for (int j = 0; j < 7; ++j) {
        int f = threadIdx.x + j * 32;
        if (f < DDIM) atomicAdd(&md[f], (float)hs[f]);
    }
}

// ---------------- fused GRU cell (6 WMMA accumulators per tile) ----------------
__global__ void gru_fused(const float* __restrict__ m, const __bf16* __restrict__ hin,
                          const __bf16* __restrict__ packIH, const __bf16* __restrict__ packHH,
                          const float* __restrict__ b_ih, const float* __restrict__ b_hh,
                          __bf16* __restrict__ hout) {
    int lane = threadIdx.x;
    int t = blockIdx.x * blockDim.y + threadIdx.y;
    int tiles = (NNODES / 16) * NT_D;
    if (t >= tiles) return;
    int mt = t / NT_D, nt = t % NT_D;
    v8f air = {}, aiz = {}, ain = {}, ahr = {}, ahz = {}, ahn = {};
    if (nt < 13) {
#pragma unroll
        for (int kt = 0; kt < KT; ++kt) {
            v16bf am = load_a_f32pad(m, DP, mt * 16, kt * 32, lane);
            v16bf ah = load_a_bf16(hin, DP, mt * 16, kt * 32, lane);
            air = wmma_bf16(am, load_b(packIH, KT, nt,           kt, lane), air);
            aiz = wmma_bf16(am, load_b(packIH, KT, nt + NT_D,    kt, lane), aiz);
            ain = wmma_bf16(am, load_b(packIH, KT, nt + 2*NT_D,  kt, lane), ain);
            ahr = wmma_bf16(ah, load_b(packHH, KT, nt,           kt, lane), ahr);
            ahz = wmma_bf16(ah, load_b(packHH, KT, nt + NT_D,    kt, lane), ahz);
            ahn = wmma_bf16(ah, load_b(packHH, KT, nt + 2*NT_D,  kt, lane), ahn);
        }
    }
    int col = nt * 16 + (lane & 15);
    bool valid = (col < DDIM);
    int colc = valid ? col : (DDIM - 1);   // clamped so bias loads stay in-bounds
    float bir = b_ih[colc];
    float biz = b_ih[DDIM + colc];
    float bin_ = b_ih[2 * DDIM + colc];
    float bhr = b_hh[colc];
    float bhz = b_hh[DDIM + colc];
    float bhn = b_hh[2 * DDIM + colc];
#pragma unroll
    for (int r = 0; r < 8; ++r) {
        int row = mt * 16 + (lane >> 4) * 8 + r;
        float ir = air[r] + bir, iz = aiz[r] + biz, in_ = ain[r] + bin_;
        float hr = ahr[r] + bhr, hz = ahz[r] + bhz, hn = ahn[r] + bhn;
        float rg = 1.0f / (1.0f + __expf(-(ir + hr)));
        float zg = 1.0f / (1.0f + __expf(-(iz + hz)));
        float ng = tanhf(in_ + rg * hn);
        float hold = (float)hin[(size_t)row * DP + col];  // in-bounds for all 224 cols
        float outv = (1.0f - zg) * ng + zg * hold;
        outv = valid ? outv : 0.0f;        // branch-free; pad cols stay 0
        hout[(size_t)row * DP + col] = (__bf16)outv;
    }
}

// ---------------- pooling: pooled[g] = max over nodes of relu(h) ----------------
__global__ void pool_max(const __bf16* __restrict__ h, const int* __restrict__ batch,
                         float* __restrict__ pooled) {
    int node = blockIdx.x * blockDim.y + threadIdx.y;
    if (node >= NNODES) return;
    int g = batch[node];
#pragma unroll
    for (int j = 0; j < 7; ++j) {
        int f = threadIdx.x + j * 32;
        if (f < DDIM) {
            float v = (float)h[(size_t)node * DP + f];
            v = v > 0.0f ? v : 0.0f;  // relu; monotone -> commutes with max
            atomicMax((unsigned*)&pooled[(size_t)g * DP + f], __float_as_uint(v));
        }
    }
}

// ---------------- final FC: out[G,2] ----------------
__global__ void fc_kernel(const float* __restrict__ pooled, const float* __restrict__ fc_w,
                          const float* __restrict__ fc_b, float* __restrict__ out) {
    int g = blockIdx.x * blockDim.x + threadIdx.x;
    if (g >= GG) return;
    float a0 = fc_b[0], a1 = fc_b[1];
    const float* p = pooled + (size_t)g * DP;
    for (int d = 0; d < DDIM; ++d) {
        float pv = p[d];
        a0 += pv * fc_w[d * 2 + 0];
        a1 += pv * fc_w[d * 2 + 1];
    }
    out[g * 2 + 0] = a0;
    out[g * 2 + 1] = a1;
}

// ---------------- host orchestration ----------------
extern "C" void kernel_launch(void* const* d_in, const int* in_sizes, int n_in,
                              void* d_out, int out_size, void* d_ws, size_t ws_size,
                              hipStream_t stream) {
    const float* x      = (const float*)d_in[0];
    const int*   ei     = (const int*)d_in[1];
    const int*   batch  = (const int*)d_in[2];
    const float* proj_w = (const float*)d_in[3];
    const float* proj_b = (const float*)d_in[4];
    const float* ggc_w  = (const float*)d_in[5];
    const float* w_ih   = (const float*)d_in[6];
    const float* w_hh   = (const float*)d_in[7];
    const float* b_ih   = (const float*)d_in[8];
    const float* b_hh   = (const float*)d_in[9];
    const float* fc_w   = (const float*)d_in[10];
    const float* fc_b   = (const float*)d_in[11];
    float* out = (float*)d_out;

    char* ws = (char*)d_ws;
    size_t off = 0;
    auto alloc = [&](size_t bytes) -> char* {
        char* p = ws + off;
        off = (off + bytes + 255) & ~(size_t)255;
        return p;
    };

    __bf16* xb      = (__bf16*)alloc((size_t)NNODES * DP * 2);
    __bf16* h0      = (__bf16*)alloc((size_t)NNODES * DP * 2);
    __bf16* h1      = (__bf16*)alloc((size_t)NNODES * DP * 2);
    __bf16* hw      = (__bf16*)alloc((size_t)NNODES * DP * 2);
    float*  m       = (float*)alloc((size_t)NNODES * DP * 4);
    __bf16* packPr  = (__bf16*)alloc((size_t)NT_D * KT * 512 * 2);
    __bf16* packG0  = (__bf16*)alloc((size_t)NT_D * KT * 512 * 2);
    __bf16* packG1  = (__bf16*)alloc((size_t)NT_D * KT * 512 * 2);
    __bf16* packIH  = (__bf16*)alloc((size_t)GATES_NT * KT * 512 * 2);
    __bf16* packHH  = (__bf16*)alloc((size_t)GATES_NT * KT * 512 * 2);
    float*  pooled  = (float*)alloc((size_t)GG * DP * 4);

    // --- pack weights to WMMA B-fragment layout (bf16) ---
    {
        int tot = NT_D * KT * 512;
        pack_b_kernel<<<(tot + 255) / 256, 256, 0, stream>>>(proj_w, packPr, KT, NT_D, INF_, DDIM, 0);
        pack_b_kernel<<<(tot + 255) / 256, 256, 0, stream>>>(ggc_w, packG0, KT, NT_D, DDIM, DDIM, 0);
        pack_b_kernel<<<(tot + 255) / 256, 256, 0, stream>>>(ggc_w + DDIM * DDIM, packG1, KT, NT_D, DDIM, DDIM, 0);
        int totg = GATES_NT * KT * 512;
        pack_b_kernel<<<(totg + 255) / 256, 256, 0, stream>>>(w_ih, packIH, KT, GATES_NT, DDIM, 3 * DP, 1);
        pack_b_kernel<<<(totg + 255) / 256, 256, 0, stream>>>(w_hh, packHH, KT, GATES_NT, DDIM, 3 * DP, 1);
    }

    // --- x -> padded bf16 ---
    {
        dim3 cblk(32, 8);
        convert_pad_x<<<(NNODES + 7) / 8, cblk, 0, stream>>>(x, xb);
    }

    int tiles = (NNODES / 16) * NT_D;
    dim3 gblk(32, 4);
    int gemmBlocks = (tiles + 3) / 4;

    // --- projection + relu ---
    proj_gemm<<<gemmBlocks, gblk, 0, stream>>>(xb, packPr, proj_b, h0);

    // --- 2 GatedGraphConv layers ---
    for (int l = 0; l < 2; ++l) {
        const __bf16* hin = (l == 0) ? h0 : h1;
        __bf16* hout      = (l == 0) ? h1 : h0;
        const __bf16* pg  = (l == 0) ? packG0 : packG1;

        ggc_gemm<<<gemmBlocks, gblk, 0, stream>>>(hin, pg, hw);

        long mElems = (long)NNODES * DP;
        fill_f32<<<(unsigned)((mElems + 255) / 256), 256, 0, stream>>>(m, 0.0f, mElems);

        dim3 sblk(32, 8);
        edge_scatter<<<(EEDGES + 7) / 8, sblk, 0, stream>>>(hw, ei, m);

        gru_fused<<<gemmBlocks, gblk, 0, stream>>>(m, hin, packIH, packHH, b_ih, b_hh, hout);
    }

    // --- pooling + FC (final h lives in h0 after layer 1) ---
    long pElems = (long)GG * DP;
    fill_f32<<<(unsigned)((pElems + 255) / 256), 256, 0, stream>>>(pooled, 0.0f, pElems);
    dim3 pblk(32, 8);
    pool_max<<<(NNODES + 7) / 8, pblk, 0, stream>>>(h0, batch, pooled);
    fc_kernel<<<1, 256, 0, stream>>>(pooled, fc_w, fc_b, out);
}
